// LlamaSdpaAttention_21990232556083
// MI455X (gfx1250) — matrix operand order repaired and checked by reading, not compiled, not run on hardware
//
#include <hip/hip_runtime.h>
#include <hip/hip_bf16.h>
#include <math.h>

// ---------------------------------------------------------------------------
// Types for CDNA5 WMMA (wave32): v16bf A/B fragments, v8f accumulator.
// ---------------------------------------------------------------------------
typedef __attribute__((ext_vector_type(16))) __bf16 v16bf;
typedef __attribute__((ext_vector_type(8)))  float  v8f;
typedef __attribute__((ext_vector_type(4)))  unsigned int u32x4;

union FragBF { v16bf v; u32x4 q[2]; };

__device__ __forceinline__ v8f wmma_bf16(v16bf a, v16bf b, v8f c) {
  // (neg_a, A, neg_b, B, c_mod, C, reuse_a, reuse_b)
  return __builtin_amdgcn_wmma_f32_16x16x32_bf16(false, a, false, b, (short)0, c,
                                                 false, false);
}

// ---------------------------------------------------------------------------
// Async global->LDS path (CDNA5 GLOBAL_LOAD_ASYNC_TO_LDS_B128, ASYNCcnt).
// Builtin signature (from compiler diagnostic): param0 = AS1 pointer to
// "int __attribute__((vector_size(16)))", param1 = LDS pointer, then two
// immediate ints (offset, cpol).
// ---------------------------------------------------------------------------
#if defined(__HIP_DEVICE_COMPILE__) && __has_builtin(__builtin_amdgcn_global_load_async_to_lds_b128)
#define HAVE_ASYNC_LDS 1
typedef int v4i_vs __attribute__((vector_size(16)));
typedef __attribute__((address_space(1))) v4i_vs* gv4i_p;
typedef __attribute__((address_space(3))) v4i_vs* lv4i_p;
__device__ __forceinline__ void async_b128(const void* g, void* l) {
  __builtin_amdgcn_global_load_async_to_lds_b128((gv4i_p)g, (lv4i_p)l, 0, 0);
}
#if __has_builtin(__builtin_amdgcn_s_wait_asynccnt)
#define WAIT_ASYNC(n) __builtin_amdgcn_s_wait_asynccnt(n)
#else
#define WAIT_ASYNC(n) asm volatile("s_wait_asynccnt %0" :: "i"(n) : "memory")
#endif
#else
#define HAVE_ASYNC_LDS 0
#endif

// ---------------------------------------------------------------------------
// f32 -> bf16 conversion (grid-stride)
// ---------------------------------------------------------------------------
__global__ __launch_bounds__(256)
void cvt_f32_bf16(const float* __restrict__ s, __hip_bfloat16* __restrict__ d, size_t n) {
  size_t i = (size_t)blockIdx.x * blockDim.x + threadIdx.x;
  size_t stride = (size_t)gridDim.x * blockDim.x;
  for (; i < n; i += stride) d[i] = __float2bfloat16(s[i]);
}

// ---------------------------------------------------------------------------
// LDS fragment readers.  LDS tiles are 128 rows x 64 cols bf16 with 8-element
// row padding: row stride = 72 elems = 144 B (16B aligned, 36-bank stride ->
// conflict-free b128 reads across 16 lanes).
// A frag (16x32 MxK): lane L<16: row M=L, K chunks [kk..kk+7],[kk+16..kk+23];
//                     lane L>=16: row M=L-16, chunks shifted by 8.
// B frag (32x16 KxN): lane L: column N=(L&15) -> tile row, same K chunks as A.
// ---------------------------------------------------------------------------
#define LDS_STRIDE 72

__device__ __forceinline__ v16bf frag_a_lds(const unsigned short* __restrict__ base,
                                            int m0, int kk, int lane) {
  const int row = m0 + (lane & 15);
  const int col = kk + (lane >> 4) * 8;
  const unsigned short* p = base + row * LDS_STRIDE + col;
  FragBF f;
  f.q[0] = *(const u32x4*)(p);
  f.q[1] = *(const u32x4*)(p + 16);
  return f.v;
}

__device__ __forceinline__ v16bf frag_b_lds(const unsigned short* __restrict__ base,
                                            int n0, int kk, int lane) {
  const int row = n0 + (lane & 15);
  const int col = kk + (lane >> 4) * 8;
  const unsigned short* p = base + row * LDS_STRIDE + col;
  FragBF f;
  f.q[0] = *(const u32x4*)(p);
  f.q[1] = *(const u32x4*)(p + 16);
  return f.v;
}

// ---------------------------------------------------------------------------
// NT GEMM:  C[m,n] = alpha * sum_k A[m,k] * B[n,k]
// Block 128x128, 8 waves (32x64 each = 2x4 WMMA tiles), K staged through LDS
// in 64-wide double-buffered stages.  M,N implied by grid (multiples of 128).
// ---------------------------------------------------------------------------
template<bool OUT_F32>
__global__ __launch_bounds__(256)
void gemm_bf16_nt(const __hip_bfloat16* __restrict__ A,
                  const __hip_bfloat16* __restrict__ Bw,
                  void* __restrict__ Cp,
                  int K, int lda, int ldb, int ldc, float alpha)
{
  __shared__ __align__(16) unsigned short As[2][128][LDS_STRIDE];
  __shared__ __align__(16) unsigned short Bs[2][128][LDS_STRIDE];

  const int lane = threadIdx.x;                 // 0..31
  const int wave = threadIdx.y;                 // 0..7
  const int tid  = wave * 32 + lane;            // 0..255
  const int wm = wave & 3;                      // 4 waves along M
  const int wn = wave >> 2;                     // 2 waves along N
  const int bm = blockIdx.y * 128;
  const int bn = blockIdx.x * 128;
  const int wrow = wm * 32;                     // wave M offset in tile
  const int wcol = wn * 64;                     // wave N offset in tile

  // Copy mapping: thread pair (2t, 2t+1) covers one 64-element row of a tile.
  const int crow = tid >> 1;                    // 0..127
  const int cseg = (tid & 1) * 32;              // 0 or 32
  const __hip_bfloat16* gA = A  + (size_t)(bm + crow) * lda + cseg;
  const __hip_bfloat16* gB = Bw + (size_t)(bn + crow) * ldb + cseg;

  v8f acc[2][4] = {};
  const int nstages = K >> 6;                   // K / 64

#if HAVE_ASYNC_LDS
  auto stage_copy = [&](int buf, int k0) {
    const __hip_bfloat16* pa = gA + k0;
    const __hip_bfloat16* pb = gB + k0;
    unsigned short* la = &As[buf][crow][cseg];
    unsigned short* lb = &Bs[buf][crow][cseg];
    #pragma unroll
    for (int i = 0; i < 4; ++i) async_b128(pa + 8 * i, la + 8 * i);
    #pragma unroll
    for (int i = 0; i < 4; ++i) async_b128(pb + 8 * i, lb + 8 * i);
  };

  stage_copy(0, 0);
  for (int s = 0; s < nstages; ++s) {
    const int cur = s & 1;
    if (s + 1 < nstages) {
      stage_copy(cur ^ 1, (s + 1) << 6);
      WAIT_ASYNC(8);          // current stage's 8 ops landed; next 8 in flight
    } else {
      WAIT_ASYNC(0);
    }
    __syncthreads();
    const unsigned short* aB = &As[cur][0][0];
    const unsigned short* bB = &Bs[cur][0][0];
    #pragma unroll
    for (int kk = 0; kk < 64; kk += 32) {
      v16bf a0 = frag_a_lds(aB, wrow,      kk, lane);
      v16bf a1 = frag_a_lds(aB, wrow + 16, kk, lane);
      v16bf b0 = frag_b_lds(bB, wcol,      kk, lane);
      v16bf b1 = frag_b_lds(bB, wcol + 16, kk, lane);
      v16bf b2 = frag_b_lds(bB, wcol + 32, kk, lane);
      v16bf b3 = frag_b_lds(bB, wcol + 48, kk, lane);
      acc[0][0] = wmma_bf16(a0, b0, acc[0][0]);
      acc[0][1] = wmma_bf16(a0, b1, acc[0][1]);
      acc[0][2] = wmma_bf16(a0, b2, acc[0][2]);
      acc[0][3] = wmma_bf16(a0, b3, acc[0][3]);
      acc[1][0] = wmma_bf16(a1, b0, acc[1][0]);
      acc[1][1] = wmma_bf16(a1, b1, acc[1][1]);
      acc[1][2] = wmma_bf16(a1, b2, acc[1][2]);
      acc[1][3] = wmma_bf16(a1, b3, acc[1][3]);
    }
    __syncthreads();
  }
#else
  // Fallback: register-staged synchronous copies, still double-buffered.
  u32x4 ra[4], rb[4];
  auto load_regs = [&](int k0) {
    const __hip_bfloat16* pa = gA + k0;
    const __hip_bfloat16* pb = gB + k0;
    #pragma unroll
    for (int i = 0; i < 4; ++i) ra[i] = *(const u32x4*)(pa + 8 * i);
    #pragma unroll
    for (int i = 0; i < 4; ++i) rb[i] = *(const u32x4*)(pb + 8 * i);
  };
  auto store_lds = [&](int buf) {
    unsigned short* la = &As[buf][crow][cseg];
    unsigned short* lb = &Bs[buf][crow][cseg];
    #pragma unroll
    for (int i = 0; i < 4; ++i) *(u32x4*)(la + 8 * i) = ra[i];
    #pragma unroll
    for (int i = 0; i < 4; ++i) *(u32x4*)(lb + 8 * i) = rb[i];
  };

  load_regs(0);
  store_lds(0);
  __syncthreads();
  for (int s = 0; s < nstages; ++s) {
    const int cur = s & 1;
    if (s + 1 < nstages) load_regs((s + 1) << 6);
    const unsigned short* aB = &As[cur][0][0];
    const unsigned short* bB = &Bs[cur][0][0];
    #pragma unroll
    for (int kk = 0; kk < 64; kk += 32) {
      v16bf a0 = frag_a_lds(aB, wrow,      kk, lane);
      v16bf a1 = frag_a_lds(aB, wrow + 16, kk, lane);
      v16bf b0 = frag_b_lds(bB, wcol,      kk, lane);
      v16bf b1 = frag_b_lds(bB, wcol + 16, kk, lane);
      v16bf b2 = frag_b_lds(bB, wcol + 32, kk, lane);
      v16bf b3 = frag_b_lds(bB, wcol + 48, kk, lane);
      acc[0][0] = wmma_bf16(a0, b0, acc[0][0]);
      acc[0][1] = wmma_bf16(a0, b1, acc[0][1]);
      acc[0][2] = wmma_bf16(a0, b2, acc[0][2]);
      acc[0][3] = wmma_bf16(a0, b3, acc[0][3]);
      acc[1][0] = wmma_bf16(a1, b0, acc[1][0]);
      acc[1][1] = wmma_bf16(a1, b1, acc[1][1]);
      acc[1][2] = wmma_bf16(a1, b2, acc[1][2]);
      acc[1][3] = wmma_bf16(a1, b3, acc[1][3]);
    }
    __syncthreads();
    if (s + 1 < nstages) {
      store_lds(cur ^ 1);
      __syncthreads();
    }
  }
#endif

  // C/D layout: lane holds column n = (lane&15); VGPR r is row r + 8*(lane>>4).
  const int nloc = lane & 15;
  const int moff = (lane >> 4) * 8;
  #pragma unroll
  for (int i = 0; i < 2; ++i) {
    #pragma unroll
    for (int j = 0; j < 4; ++j) {
      const int mbase = bm + wrow + 16 * i + moff;
      const int n     = bn + wcol + 16 * j + nloc;
      #pragma unroll
      for (int r = 0; r < 8; ++r) {
        float vv = acc[i][j][r] * alpha;
        if (OUT_F32)
          ((float*)Cp)[(size_t)(mbase + r) * ldc + n] = vv;
        else
          ((__hip_bfloat16*)Cp)[(size_t)(mbase + r) * ldc + n] = __float2bfloat16(vv);
      }
    }
  }
}

// ---------------------------------------------------------------------------
// RoPE in-place on Q and K (bf16, [B*S, D]).  One thread per (row, j<D/2) pair.
// ---------------------------------------------------------------------------
__global__ __launch_bounds__(256)
void rope_inplace(__hip_bfloat16* __restrict__ Q, __hip_bfloat16* __restrict__ Kc,
                  int S, int Dm, size_t total)
{
  const int h = Dm >> 1;
  size_t idx = (size_t)blockIdx.x * blockDim.x + threadIdx.x;
  if (idx >= total) return;
  size_t bs = idx / h;
  int j = (int)(idx - bs * h);
  int s = (int)(bs % S);
  // inv_freq[j] = theta^(-(2j)/D) = exp(-ln(theta) * 2j/D)
  float invf = __expf(-9.210340371976184f * (2.0f * (float)j) / (float)Dm);
  float ang  = (float)s * invf;
  float sn, cs;
  __sincosf(ang, &sn, &cs);
  size_t i1 = bs * (size_t)Dm + j;
  size_t i2 = i1 + h;
  float q1 = __bfloat162float(Q[i1]), q2 = __bfloat162float(Q[i2]);
  Q[i1] = __float2bfloat16(q1 * cs - q2 * sn);
  Q[i2] = __float2bfloat16(q2 * cs + q1 * sn);
  float k1 = __bfloat162float(Kc[i1]), k2 = __bfloat162float(Kc[i2]);
  Kc[i1] = __float2bfloat16(k1 * cs - k2 * sn);
  Kc[i2] = __float2bfloat16(k2 * cs + k1 * sn);
}

// ---------------------------------------------------------------------------
// Tiled bf16 transpose (per batch via blockIdx.z): dst[d][s] = src[s][d]
// ---------------------------------------------------------------------------
__global__ __launch_bounds__(256)
void transpose_bf16(const __hip_bfloat16* __restrict__ src,
                    __hip_bfloat16* __restrict__ dst, int rows, int cols)
{
  __shared__ unsigned short tile[32][33];
  const size_t batch = (size_t)blockIdx.z * rows * cols;
  const unsigned short* s = (const unsigned short*)src + batch;
  unsigned short* d = (unsigned short*)dst + batch;
  int x  = blockIdx.x * 32 + threadIdx.x;
  int y0 = blockIdx.y * 32;
  #pragma unroll
  for (int i = threadIdx.y; i < 32; i += 8)
    tile[i][threadIdx.x] = s[(size_t)(y0 + i) * cols + x];
  __syncthreads();
  int ox  = blockIdx.y * 32 + threadIdx.x;
  int oy0 = blockIdx.x * 32;
  #pragma unroll
  for (int i = threadIdx.y; i < 32; i += 8)
    d[(size_t)(oy0 + i) * rows + ox] = tile[threadIdx.x][i];
}

// ---------------------------------------------------------------------------
// Row softmax: f32 scores [rows, ncols] -> bf16 probs.  One 256-thread block
// per row; ncols == 2048 == 256*8.
// ---------------------------------------------------------------------------
__global__ __launch_bounds__(256)
void softmax_rows(const float* __restrict__ Sc, __hip_bfloat16* __restrict__ P, int ncols)
{
  __shared__ float red[8];
  const int tid  = threadIdx.x;
  const int lane = tid & 31, wave = tid >> 5;
  const size_t row = blockIdx.x;
  const float* src = Sc + row * (size_t)ncols;
  __hip_bfloat16* dst = P + row * (size_t)ncols;

  float v[8];
  float mx = -3.4e38f;
  #pragma unroll
  for (int i = 0; i < 8; ++i) { v[i] = src[tid + 256 * i]; mx = fmaxf(mx, v[i]); }
  #pragma unroll
  for (int o = 16; o > 0; o >>= 1) mx = fmaxf(mx, __shfl_xor(mx, o, 32));
  if (lane == 0) red[wave] = mx;
  __syncthreads();
  mx = red[0];
  #pragma unroll
  for (int i = 1; i < 8; ++i) mx = fmaxf(mx, red[i]);
  __syncthreads();

  float sum = 0.f;
  #pragma unroll
  for (int i = 0; i < 8; ++i) { v[i] = __expf(v[i] - mx); sum += v[i]; }
  #pragma unroll
  for (int o = 16; o > 0; o >>= 1) sum += __shfl_xor(sum, o, 32);
  if (lane == 0) red[wave] = sum;
  __syncthreads();
  sum = 0.f;
  #pragma unroll
  for (int i = 0; i < 8; ++i) sum += red[i];
  float inv = 1.0f / sum;
  #pragma unroll
  for (int i = 0; i < 8; ++i) dst[tid + 256 * i] = __float2bfloat16(v[i] * inv);
}

// ---------------------------------------------------------------------------
// Launch pipeline
// ---------------------------------------------------------------------------
extern "C" void kernel_launch(void* const* d_in, const int* in_sizes, int n_in,
                              void* d_out, int out_size, void* d_ws, size_t ws_size,
                              hipStream_t stream)
{
  (void)in_sizes; (void)n_in; (void)out_size; (void)ws_size;
  constexpr int Bb = 4, S = 2048, Dm = 2048;
  constexpr size_t MS = (size_t)Bb * S;          // 8192 rows
  const float* x  = (const float*)d_in[0];
  const float* Wq = (const float*)d_in[1];
  const float* Wk = (const float*)d_in[2];
  const float* Wv = (const float*)d_in[3];
  const float* Wo = (const float*)d_in[4];
  float* out = (float*)d_out;

  char* w = (char*)d_ws;
  size_t off = 0;
  auto take = [&](size_t bytes) { char* p = w + off; off += (bytes + 255) & ~(size_t)255; return p; };
  __hip_bfloat16* Xb  = (__hip_bfloat16*)take(MS * Dm * 2);           // X bf16 (reused as ctx)
  __hip_bfloat16* Wqb = (__hip_bfloat16*)take((size_t)Dm * Dm * 2);
  __hip_bfloat16* Wkb = (__hip_bfloat16*)take((size_t)Dm * Dm * 2);
  __hip_bfloat16* Wvb = (__hip_bfloat16*)take((size_t)Dm * Dm * 2);
  __hip_bfloat16* Wob = (__hip_bfloat16*)take((size_t)Dm * Dm * 2);
  __hip_bfloat16* Qb  = (__hip_bfloat16*)take(MS * Dm * 2);
  __hip_bfloat16* Kb  = (__hip_bfloat16*)take(MS * Dm * 2);
  __hip_bfloat16* Vb  = (__hip_bfloat16*)take(MS * Dm * 2);
  __hip_bfloat16* VTb = (__hip_bfloat16*)take(MS * Dm * 2);
  float*          Sc  = (float*)take((size_t)Bb * S * S * 4);         // f32 scores
  __hip_bfloat16* Pb  = (__hip_bfloat16*)take((size_t)Bb * S * S * 2);
  __hip_bfloat16* Ctx = Xb;  // reuse: X not needed after QKV projections

  // 1) convert inputs to bf16
  cvt_f32_bf16<<<4096, 256, 0, stream>>>(x,  Xb,  MS * Dm);
  cvt_f32_bf16<<<2048, 256, 0, stream>>>(Wq, Wqb, (size_t)Dm * Dm);
  cvt_f32_bf16<<<2048, 256, 0, stream>>>(Wk, Wkb, (size_t)Dm * Dm);
  cvt_f32_bf16<<<2048, 256, 0, stream>>>(Wv, Wvb, (size_t)Dm * Dm);
  cvt_f32_bf16<<<2048, 256, 0, stream>>>(Wo, Wob, (size_t)Dm * Dm);

  dim3 blk(32, 8);
  // 2) QKV projections: [8192,2048] x [2048,2048]^T
  dim3 gproj(Dm / 128, MS / 128);
  gemm_bf16_nt<false><<<gproj, blk, 0, stream>>>(Xb, Wqb, Qb, Dm, Dm, Dm, Dm, 1.0f);
  gemm_bf16_nt<false><<<gproj, blk, 0, stream>>>(Xb, Wkb, Kb, Dm, Dm, Dm, Dm, 1.0f);
  gemm_bf16_nt<false><<<gproj, blk, 0, stream>>>(Xb, Wvb, Vb, Dm, Dm, Dm, Dm, 1.0f);

  // 3) RoPE on Q, K
  {
    size_t total = MS * (size_t)(Dm / 2);
    rope_inplace<<<(unsigned)((total + 255) / 256), 256, 0, stream>>>(Qb, Kb, S, Dm, total);
  }

  // 4) V -> V^T per batch
  {
    dim3 gt(Dm / 32, S / 32, Bb);
    transpose_bf16<<<gt, blk, 0, stream>>>(Vb, VTb, S, Dm);
  }

  // 5) scores = (Q K^T) / sqrt(D)  (f32), per batch
  const float alpha = 1.0f / sqrtf((float)Dm);
  dim3 gattn(S / 128, S / 128);
  for (int b = 0; b < Bb; ++b) {
    gemm_bf16_nt<true><<<gattn, blk, 0, stream>>>(
        Qb + (size_t)b * S * Dm, Kb + (size_t)b * S * Dm,
        Sc + (size_t)b * S * S, Dm, Dm, Dm, S, alpha);
  }

  // 6) softmax rows -> bf16 probs
  softmax_rows<<<(unsigned)(Bb * S), 256, 0, stream>>>(Sc, Pb, S);

  // 7) ctx = P V  (uses V^T so reduction dim is contiguous), per batch
  for (int b = 0; b < Bb; ++b) {
    gemm_bf16_nt<false><<<gattn, blk, 0, stream>>>(
        Pb + (size_t)b * S * S, VTb + (size_t)b * (size_t)Dm * S,
        Ctx + (size_t)b * S * Dm, S, S, S, Dm, 1.0f);
  }

  // 8) out = ctx Wo^T  (f32 out)
  gemm_bf16_nt<true><<<gproj, blk, 0, stream>>>(Ctx, Wob, out, Dm, Dm, Dm, Dm, 1.0f);
}
